// RAEDiTAttention_1975684956823
// MI455X (gfx1250) — matrix operand order repaired and checked by reading, not compiled, not run on hardware
//
#include <hip/hip_runtime.h>
#include <hip/hip_bf16.h>

typedef __attribute__((ext_vector_type(16))) _Float16 v16h;
typedef __attribute__((ext_vector_type(8)))  _Float16 v8h;
typedef __attribute__((ext_vector_type(8)))  float    v8f;

// async-copy pointer types: builtin wants int4* in AS1 (global) / AS3 (LDS)
typedef __attribute__((__vector_size__(16))) int vsi4;
typedef __attribute__((address_space(1))) vsi4 as1_vsi4;
typedef __attribute__((address_space(3))) vsi4 as3_vsi4;

#define DIM 1536
#define NHEAD 16
#define HDIM 96
#define SEQ 1024
#define BATCH 8
#define ROWS (BATCH * SEQ)          // 8192
#define SCALE 0.10206207261596575f  // 1/sqrt(96)

// ---- CDNA5 async load-to-LDS (ASYNCcnt) support, guarded ----
#if defined(__has_builtin)
#  if __has_builtin(__builtin_amdgcn_global_load_async_to_lds_b128) && \
      __has_builtin(__builtin_amdgcn_s_wait_asynccnt)
#    define USE_ASYNC_LDS 1
#  endif
#endif
#ifndef USE_ASYNC_LDS
#  define USE_ASYNC_LDS 0
#endif

static __device__ inline v8f v8f_zero() {
    v8f z = {0.f, 0.f, 0.f, 0.f, 0.f, 0.f, 0.f, 0.f};
    return z;
}

static __device__ inline v8f wmma_f16(v16h a, v16h b, v8f c) {
    // v_wmma_f32_16x16x32_f16
    return __builtin_amdgcn_wmma_f32_16x16x32_f16(false, a, false, b, (short)0, c, false, false);
}

// A-operand (16x32, M x K) from row-major f16 [*, ld]:
// lane m (0..15): VGPR0-7 = K[k0..k0+15]
// lane m+16:      VGPR0-7 = K[k0+16..k0+31]
static __device__ inline v16h load_A(const _Float16* base, int row0, int k0, int ld, int lane) {
    int m = lane & 15, h = lane >> 4;
    const _Float16* p = base + (size_t)(row0 + m) * ld + k0 + h * 16;
    union { v16h v; v8h h8[2]; } u;
    u.h8[0] = *(const v8h*)(p);
    u.h8[1] = *(const v8h*)(p + 8);
    return u.v;
}

// B-operand (32x16, K x N) where B[k][n] = Wrows[n0+n][k0+k] (i.e. B = W^T tile):
// lanes 0-15 hold K = k0..k0+15 for column n=lane, lanes 16-31 hold K = k0+16..+31.
static __device__ inline v16h load_B_rows(const _Float16* Wrows, int n0, int k0, int ld, int lane) {
    int n = lane & 15, h = lane >> 4;
    const _Float16* p = Wrows + (size_t)(n0 + n) * ld + k0 + h * 16;
    return *(const v16h*)p;  // 32B contiguous, 32B aligned
}

// B-operand from an LDS row base: 16 contiguous halves starting at `off` (16B aligned)
static __device__ inline v16h load_B_lds(const _Float16* rowbase, int off) {
    union { v16h v; v8h h8[2]; } u;
    u.h8[0] = *(const v8h*)(rowbase + off);
    u.h8[1] = *(const v8h*)(rowbase + off + 8);
    return u.v;
}

// copy one 16B chunk global -> LDS (async on CDNA5, sync fallback otherwise)
static __device__ inline void cp16_g2l(const _Float16* src, _Float16* dst) {
#if USE_ASYNC_LDS
    __builtin_amdgcn_global_load_async_to_lds_b128(
        (as1_vsi4*)src, (as3_vsi4*)dst, 0, 0);
#else
    *(v8h*)dst = *(const v8h*)src;
#endif
}

// ---------------- small prep kernels ----------------

__global__ void rope_table_kernel(float* __restrict__ cosT, float* __restrict__ sinT) {
    int i = blockIdx.x * blockDim.x + threadIdx.x;
    if (i >= SEQ * HDIM) return;
    int s = i / HDIM, d = i % HDIM;
    int r = s >> 5, c = s & 31;
    int dd = (d < 48) ? d : d - 48;
    int j = dd >> 1;  // 0..23
    float pos = (float)((d < 48) ? r : c);
    // f = 10000^(-(2j)/48) = exp(-ln(10000) * 2j/48)
    float f = __expf(-(float)(2 * j) * (9.210340371976184f / 48.0f));
    float a = pos * f;
    cosT[i] = __cosf(a);
    sinT[i] = __sinf(a);
}

__global__ void cvt_f32_f16_kernel(const float* __restrict__ in, _Float16* __restrict__ out, int n) {
    for (int i = blockIdx.x * blockDim.x + threadIdx.x; i < n; i += gridDim.x * blockDim.x)
        out[i] = (_Float16)in[i];
}

// ---------------- fused QKV projection + bias + RMSNorm + RoPE ----------------
// grid: (ROWS/128, NHEAD, 3), block: 128 (4 waves, each a 32-row block)
__global__ __launch_bounds__(128) void qkv_kernel(
    const _Float16* __restrict__ Xh,
    const _Float16* __restrict__ Wqh, const _Float16* __restrict__ Wkh, const _Float16* __restrict__ Wvh,
    const float* __restrict__ bq, const float* __restrict__ bk, const float* __restrict__ bv,
    const float* __restrict__ qnw, const float* __restrict__ knw,
    const float* __restrict__ cosT, const float* __restrict__ sinT,
    _Float16* __restrict__ Qh, _Float16* __restrict__ Kh, _Float16* __restrict__ Vth) {
    const int lane = threadIdx.x & 31;
    const int w = threadIdx.x >> 5;
    const int row0 = (blockIdx.x * 4 + w) * 32;
    const int head = blockIdx.y;
    const int mat = blockIdx.z;  // 0=Q, 1=K, 2=V

    const _Float16* W = (mat == 0) ? Wqh : (mat == 1) ? Wkh : Wvh;
    const float* bias = (mat == 0) ? bq : (mat == 1) ? bk : bv;

    v8f acc[2][6];
#pragma unroll
    for (int rr = 0; rr < 2; ++rr)
#pragma unroll
        for (int t = 0; t < 6; ++t) acc[rr][t] = v8f_zero();

    for (int k0 = 0; k0 < DIM; k0 += 32) {
        v16h a0 = load_A(Xh, row0, k0, DIM, lane);
        v16h a1 = load_A(Xh, row0 + 16, k0, DIM, lane);
#pragma unroll
        for (int t = 0; t < 6; ++t) {
            v16h bm = load_B_rows(W, head * HDIM + t * 16, k0, DIM, lane);
            acc[0][t] = wmma_f16(a0, bm, acc[0][t]);
            acc[1][t] = wmma_f16(a1, bm, acc[1][t]);
        }
    }

    const int g = lane >> 4, nn = lane & 15;

#pragma unroll
    for (int rr = 0; rr < 2; ++rr) {
        const int rbase = row0 + rr * 16;
        // bias (per output column, broadcast over rows)
#pragma unroll
        for (int t = 0; t < 6; ++t) {
            float bb = bias[head * HDIM + t * 16 + nn];
#pragma unroll
            for (int r = 0; r < 8; ++r) acc[rr][t][r] += bb;
        }

        if (mat < 2) {
            const float* nw = (mat == 0) ? qnw : knw;
#pragma unroll
            for (int r = 0; r < 8; ++r) {
                // RMS over the 96 dims of this head (row = r + 8*g)
                float ss = 0.f;
#pragma unroll
                for (int t = 0; t < 6; ++t) { float v = acc[rr][t][r]; ss += v * v; }
                ss += __shfl_xor(ss, 1, 32);
                ss += __shfl_xor(ss, 2, 32);
                ss += __shfl_xor(ss, 4, 32);
                ss += __shfl_xor(ss, 8, 32);
                float inv = rsqrtf(ss * (1.0f / HDIM) + 1e-6f);
                int stok = (rbase + r + 8 * g) & (SEQ - 1);
#pragma unroll
                for (int t = 0; t < 6; ++t) {
                    int d = t * 16 + nn;
                    float v = acc[rr][t][r] * inv * nw[d];
                    float partner = __shfl_xor(v, 1, 32);  // value at d^1
                    float cc = cosT[stok * HDIM + d];
                    float sn = sinT[stok * HDIM + d];
                    float sign = (d & 1) ? 1.f : -1.f;  // x_rot[2i]=-x[2i+1], x_rot[2i+1]=x[2i]
                    acc[rr][t][r] = v * cc + sign * partner * sn;
                }
            }
        }

        // store: Q,K -> [b,h,s,d] f16 ; V -> transposed [b,h,d,s] f16
#pragma unroll
        for (int r = 0; r < 8; ++r) {
            int rg = rbase + r + 8 * g;
            int b = rg >> 10, s = rg & (SEQ - 1);
            size_t bh = (size_t)b * NHEAD + head;
#pragma unroll
            for (int t = 0; t < 6; ++t) {
                int d = t * 16 + nn;
                _Float16 hv = (_Float16)acc[rr][t][r];
                if (mat == 0)      Qh[(bh * SEQ + s) * HDIM + d] = hv;
                else if (mat == 1) Kh[(bh * SEQ + s) * HDIM + d] = hv;
                else               Vth[(bh * HDIM + d) * SEQ + s] = hv;
            }
        }
    }
}

// cooperative stage of one 32-key K tile (32x96 f16) + V tile (96x32 f16) into LDS
// 6 x 16B chunks per thread (128 threads)
static __device__ inline void stage_kv(const _Float16* __restrict__ Khead,
                                       const _Float16* __restrict__ Vhead,
                                       int key0, _Float16* Kt, _Float16* Vt, int tid) {
    // K tile: 32 rows x 192B (12 chunks/row)
    {
        int krow = tid & 31;
        int cb = tid >> 5;  // 0..3
#pragma unroll
        for (int j = 0; j < 3; ++j) {
            int col = cb + j * 4;  // 0..11
            cp16_g2l(Khead + (size_t)(key0 + krow) * HDIM + col * 8,
                     Kt + krow * HDIM + col * 8);
        }
    }
    // V tile: 96 rows x 64B (4 chunks/row)
#pragma unroll
    for (int j = 0; j < 3; ++j) {
        int c = tid + j * 128;  // 0..383
        int vrow = c >> 2, vcol = c & 3;
        cp16_g2l(Vhead + (size_t)vrow * SEQ + key0 + vcol * 8,
                 Vt + vrow * 32 + vcol * 8);
    }
}

// ---------------- flash attention (online softmax), per wave: 16 q rows ----------------
// grid: (SEQ/64, NHEAD, BATCH), block: 128 (4 waves share b,h -> share K/V tiles in LDS)
__global__ __launch_bounds__(128) void attn_kernel(
    const _Float16* __restrict__ Qh, const _Float16* __restrict__ Kh,
    const _Float16* __restrict__ Vth, _Float16* __restrict__ Oh) {
    __shared__ _Float16 Kt[2][32][HDIM];   // [buf][key][d]   12 KB
    __shared__ _Float16 Vt[2][HDIM][32];   // [buf][d][key]   12 KB
    __shared__ _Float16 plds[4][16][32];   // per-wave probs staging (C-layout -> A-layout)

    const int tid = threadIdx.x;
    const int lane = tid & 31;
    const int w = tid >> 5;
    const int h = blockIdx.y, b = blockIdx.z;
    const int q0 = (blockIdx.x * 4 + w) * 16;
    const size_t bh = (size_t)b * NHEAD + h;

    const _Float16* Qhead = Qh + bh * SEQ * HDIM;
    const _Float16* Khead = Kh + bh * SEQ * HDIM;
    const _Float16* Vhead = Vth + bh * HDIM * SEQ;

    v16h aq[3];
#pragma unroll
    for (int c = 0; c < 3; ++c) aq[c] = load_A(Qhead, q0, c * 32, HDIM, lane);

    v8f o[6];
#pragma unroll
    for (int t = 0; t < 6; ++t) o[t] = v8f_zero();
    float mrow[8], lrow[8];
#pragma unroll
    for (int r = 0; r < 8; ++r) { mrow[r] = -1e30f; lrow[r] = 0.f; }

    const int g = lane >> 4, nn = lane & 15;

    // prologue: stage tiles for the first key block
    stage_kv(Khead, Vhead, 0, &Kt[0][0][0], &Vt[0][0][0], tid);

    for (int it = 0; it < SEQ / 32; ++it) {
        const int buf = it & 1;
        // prefetch next tiles into the other buffer while waiting on this one
        if (it + 1 < SEQ / 32) {
            stage_kv(Khead, Vhead, (it + 1) * 32, &Kt[buf ^ 1][0][0], &Vt[buf ^ 1][0][0], tid);
#if USE_ASYNC_LDS
            __builtin_amdgcn_s_wait_asynccnt(6);  // current tile's 6 chunks complete
#endif
        } else {
#if USE_ASYNC_LDS
            __builtin_amdgcn_s_wait_asynccnt(0);
#endif
        }
        __syncthreads();

        // scores: S = Q (16x96) x K^T (96 x 32), two 16-col WMMA tiles, K operand from LDS
        v8f s0 = v8f_zero(), s1 = v8f_zero();
#pragma unroll
        for (int c = 0; c < 3; ++c) {
            v16h b0 = load_B_lds(&Kt[buf][nn][0], c * 32 + g * 16);
            v16h b1 = load_B_lds(&Kt[buf][16 + nn][0], c * 32 + g * 16);
            s0 = wmma_f16(aq[c], b0, s0);
            s1 = wmma_f16(aq[c], b1, s1);
        }

        // online softmax over this 16x32 score block
#pragma unroll
        for (int r = 0; r < 8; ++r) {
            float x0 = s0[r] * SCALE, x1 = s1[r] * SCALE;
            float mx = fmaxf(x0, x1);
            mx = fmaxf(mx, __shfl_xor(mx, 1, 32));
            mx = fmaxf(mx, __shfl_xor(mx, 2, 32));
            mx = fmaxf(mx, __shfl_xor(mx, 4, 32));
            mx = fmaxf(mx, __shfl_xor(mx, 8, 32));
            float mnew = fmaxf(mrow[r], mx);
            float alpha = __expf(mrow[r] - mnew);
            float p0 = __expf(x0 - mnew);
            float p1 = __expf(x1 - mnew);
            float rs = p0 + p1;
            rs += __shfl_xor(rs, 1, 32);
            rs += __shfl_xor(rs, 2, 32);
            rs += __shfl_xor(rs, 4, 32);
            rs += __shfl_xor(rs, 8, 32);
            lrow[r] = lrow[r] * alpha + rs;
            mrow[r] = mnew;
            s0[r] = p0;
            s1[r] = p1;
#pragma unroll
            for (int t = 0; t < 6; ++t) o[t][r] *= alpha;
        }

        // transpose probs (C-layout f32) -> A-layout f16 via per-wave LDS (in-order DS)
#pragma unroll
        for (int r = 0; r < 8; ++r) {
            plds[w][g * 8 + r][nn]      = (_Float16)s0[r];
            plds[w][g * 8 + r][16 + nn] = (_Float16)s1[r];
        }
        union { v16h v; v8h h8[2]; } up;
        up.h8[0] = *(const v8h*)&plds[w][nn][g * 16];
        up.h8[1] = *(const v8h*)&plds[w][nn][g * 16 + 8];

        // O += P (16x32) x V (32 keys x 96 dims), V operand from LDS (transposed tile)
#pragma unroll
        for (int t = 0; t < 6; ++t) {
            v16h bv = load_B_lds(&Vt[buf][t * 16 + nn][0], g * 16);
            o[t] = wmma_f16(up.v, bv, o[t]);
        }
        __syncthreads();  // everyone done with buf before it is overwritten
    }

    // normalize and store to Oh as [b, s, h*96+d] f16 (row-major [8192, 1536])
#pragma unroll
    for (int r = 0; r < 8; ++r) {
        float invl = 1.0f / lrow[r];
        int s = q0 + g * 8 + r;
#pragma unroll
        for (int t = 0; t < 6; ++t) {
            int d = t * 16 + nn;
            Oh[((size_t)(b * SEQ + s)) * DIM + h * HDIM + d] = (_Float16)(o[t][r] * invl);
        }
    }
}

// ---------------- output projection: Y = Oh @ Wo^T + bo (f32 out) ----------------
// grid: (ROWS/128, DIM/96), block: 128 (4 waves, each a 32-row block)
__global__ __launch_bounds__(128) void oproj_kernel(
    const _Float16* __restrict__ Ohh, const _Float16* __restrict__ Woh,
    const float* __restrict__ bo, float* __restrict__ Y) {
    const int lane = threadIdx.x & 31;
    const int w = threadIdx.x >> 5;
    const int row0 = (blockIdx.x * 4 + w) * 32;
    const int ct = blockIdx.y;  // 96-wide column tile

    v8f acc[2][6];
#pragma unroll
    for (int rr = 0; rr < 2; ++rr)
#pragma unroll
        for (int t = 0; t < 6; ++t) acc[rr][t] = v8f_zero();

    for (int k0 = 0; k0 < DIM; k0 += 32) {
        v16h a0 = load_A(Ohh, row0, k0, DIM, lane);
        v16h a1 = load_A(Ohh, row0 + 16, k0, DIM, lane);
#pragma unroll
        for (int t = 0; t < 6; ++t) {
            v16h bm = load_B_rows(Woh, ct * HDIM + t * 16, k0, DIM, lane);
            acc[0][t] = wmma_f16(a0, bm, acc[0][t]);
            acc[1][t] = wmma_f16(a1, bm, acc[1][t]);
        }
    }

    const int g = lane >> 4, nn = lane & 15;
#pragma unroll
    for (int rr = 0; rr < 2; ++rr) {
#pragma unroll
        for (int r = 0; r < 8; ++r) {
            int row = row0 + rr * 16 + r + 8 * g;
#pragma unroll
            for (int t = 0; t < 6; ++t) {
                int col = ct * HDIM + t * 16 + nn;
                Y[(size_t)row * DIM + col] = acc[rr][t][r] + bo[col];
            }
        }
    }
}

// ---------------- host-side launch ----------------

extern "C" void kernel_launch(void* const* d_in, const int* in_sizes, int n_in,
                              void* d_out, int out_size, void* d_ws, size_t ws_size,
                              hipStream_t stream) {
    (void)in_sizes; (void)n_in; (void)out_size; (void)ws_size;
    const float* hidden = (const float*)d_in[0];
    const float* Wq = (const float*)d_in[1];
    const float* bq = (const float*)d_in[2];
    const float* Wk = (const float*)d_in[3];
    const float* bk = (const float*)d_in[4];
    const float* Wv = (const float*)d_in[5];
    const float* bv = (const float*)d_in[6];
    const float* qnw = (const float*)d_in[7];
    const float* knw = (const float*)d_in[8];
    const float* Wo = (const float*)d_in[9];
    const float* bo = (const float*)d_in[10];

    // workspace bump allocator (256B aligned)
    size_t off = 0;
    auto alloc = [&](size_t bytes) -> void* {
        void* p = (char*)d_ws + off;
        off += (bytes + 255) & ~(size_t)255;
        return p;
    };
    float* cosT = (float*)alloc((size_t)SEQ * HDIM * 4);
    float* sinT = (float*)alloc((size_t)SEQ * HDIM * 4);
    _Float16* Xh  = (_Float16*)alloc((size_t)ROWS * DIM * 2);
    _Float16* Wqh = (_Float16*)alloc((size_t)DIM * DIM * 2);
    _Float16* Wkh = (_Float16*)alloc((size_t)DIM * DIM * 2);
    _Float16* Wvh = (_Float16*)alloc((size_t)DIM * DIM * 2);
    _Float16* Woh = (_Float16*)alloc((size_t)DIM * DIM * 2);
    _Float16* Qh  = (_Float16*)alloc((size_t)ROWS * DIM * 2);
    _Float16* Kh  = (_Float16*)alloc((size_t)ROWS * DIM * 2);
    _Float16* Vth = (_Float16*)alloc((size_t)ROWS * DIM * 2);
    _Float16* Ohh = (_Float16*)alloc((size_t)ROWS * DIM * 2);

    rope_table_kernel<<<(SEQ * HDIM + 255) / 256, 256, 0, stream>>>(cosT, sinT);

    const int nX = ROWS * DIM;    // 12582912
    const int nW = DIM * DIM;     // 2359296
    cvt_f32_f16_kernel<<<4096, 256, 0, stream>>>(hidden, Xh, nX);
    cvt_f32_f16_kernel<<<2048, 256, 0, stream>>>(Wq, Wqh, nW);
    cvt_f32_f16_kernel<<<2048, 256, 0, stream>>>(Wk, Wkh, nW);
    cvt_f32_f16_kernel<<<2048, 256, 0, stream>>>(Wv, Wvh, nW);
    cvt_f32_f16_kernel<<<2048, 256, 0, stream>>>(Wo, Woh, nW);

    qkv_kernel<<<dim3(ROWS / 128, NHEAD, 3), 128, 0, stream>>>(
        Xh, Wqh, Wkh, Wvh, bq, bk, bv, qnw, knw, cosT, sinT, Qh, Kh, Vth);

    attn_kernel<<<dim3(SEQ / 64, NHEAD, BATCH), 128, 0, stream>>>(Qh, Kh, Vth, Ohh);

    oproj_kernel<<<dim3(ROWS / 128, DIM / HDIM), 128, 0, stream>>>(Ohh, Woh, bo, (float*)d_out);
}
